// DSCBR_11811160064496
// MI455X (gfx1250) — compile-verified, matrix operand order checked by reading, not run on hardware
//
#include <hip/hip_runtime.h>
#include <hip/hip_bf16.h>

typedef __attribute__((ext_vector_type(16))) __bf16 v16bf;
typedef __attribute__((ext_vector_type(8)))  float  v8f;

#define D 64

// ---------------- zero fill (vectorized, b128 stores) ----------------
__global__ void fillz_kernel(float4* __restrict__ p, long n4) {
    long i = (long)blockIdx.x * blockDim.x + threadIdx.x;
    if (i < n4) p[i] = make_float4(0.f, 0.f, 0.f, 0.f);
}

__global__ void zero2_kernel(float* o) {
    if (threadIdx.x < 2) o[threadIdx.x] = 0.0f;
}

// ---------------- COO SpMM: out[row] += scale*val*feat[col] ----------------
// 16 threads per edge; each does one b128 gather + 4 f32 scatter-atomics.
__global__ void spmm_kernel(const int* __restrict__ row, const int* __restrict__ col,
                            const float* __restrict__ val, const float* __restrict__ feat,
                            float* __restrict__ out, long nE, float scale) {
    long i = (long)blockIdx.x * blockDim.x + threadIdx.x;
    long total = nE << 4;
    if (i >= total) return;
    long e = i >> 4;
    int  q = (int)(i & 15) << 2;           // dim offset 0,4,...,60
    long r = row[e], c = col[e];
    float s = scale * val[e];
    const float4 f = *(const float4*)(feat + c * D + q);
    float* o = out + r * D + q;
    atomicAdd(o + 0, s * f.x);
    atomicAdd(o + 1, s * f.y);
    atomicAdd(o + 2, s * f.z);
    atomicAdd(o + 3, s * f.w);
}

// ---------------- acc += f / max(||f||,1e-12), one wave per row ----------------
__global__ __launch_bounds__(256) void acc_norm_kernel(float* __restrict__ acc,
                                                       const float* __restrict__ f, long nrows) {
    long wid = (long)blockIdx.x * (blockDim.x >> 5) + (threadIdx.x >> 5);
    int lane = threadIdx.x & 31;
    if (wid >= nrows) return;
    const float* fr = f + wid * D;
    float a0 = fr[lane], a1 = fr[lane + 32];
    float ss = a0 * a0 + a1 * a1;
    #pragma unroll
    for (int off = 16; off; off >>= 1) ss += __shfl_xor(ss, off, 32);
    float inv = 1.0f / fmaxf(sqrtf(ss), 1e-12f);
    acc[wid * D + lane]      += a0 * inv;
    acc[wid * D + lane + 32] += a1 * inv;
}

// ---------------- pred[i,j] = <IL_u[u],IL_b[b]> + <BL_u[u],BL_b[b]> ----------------
__global__ __launch_bounds__(128) void pred_kernel(const float* __restrict__ IL_u,
                                                   const float* __restrict__ IL_b,
                                                   const float* __restrict__ BL_u,
                                                   const float* __restrict__ BL_b,
                                                   const int* __restrict__ users,
                                                   const int* __restrict__ bundles,
                                                   float* __restrict__ pred, int batch) {
    int wid  = blockIdx.x * (blockDim.x >> 5) + (threadIdx.x >> 5);
    int lane = threadIdx.x & 31;
    if (wid >= batch * 2) return;
    int i = wid >> 1, j = wid & 1;
    long u = users[i];
    long b = bundles[i * 2 + j];
    const float* a0 = IL_u + u * D; const float* b0 = IL_b + b * D;
    const float* a1 = BL_u + u * D; const float* b1 = BL_b + b * D;
    float s = a0[lane] * b0[lane] + a0[lane + 32] * b0[lane + 32]
            + a1[lane] * b1[lane] + a1[lane + 32] * b1[lane + 32];
    #pragma unroll
    for (int off = 16; off; off >>= 1) s += __shfl_xor(s, off, 32);
    if (lane == 0) pred[wid] = s;
}

// ---------------- BPR: mean softplus(pred_neg - pred_pos) -> out[0] ----------------
__global__ __launch_bounds__(256) void bpr_kernel(const float* __restrict__ pred,
                                                  float* __restrict__ out, int batch) {
    int i = blockIdx.x * blockDim.x + threadIdx.x;
    float v = 0.0f;
    if (i < batch) {
        float x = pred[i * 2 + 1] - pred[i * 2 + 0];
        v = (x > 0.0f) ? (x + log1pf(expf(-x))) : log1pf(expf(x));
        v /= (float)batch;
    }
    __shared__ float sm[256];
    sm[threadIdx.x] = v;
    __syncthreads();
    for (int s = 128; s; s >>= 1) {
        if (threadIdx.x < s) sm[threadIdx.x] += sm[threadIdx.x + s];
        __syncthreads();
    }
    if (threadIdx.x == 0) atomicAdd(out, sm[0]);
}

// ---------------- gather + L2-normalize row -> bf16, one wave per row ----------------
__global__ __launch_bounds__(256) void gather_norm_bf16_kernel(const float* __restrict__ src,
                                                               const int* __restrict__ idx,
                                                               int idx_stride,
                                                               __bf16* __restrict__ dst, int nrows) {
    int wid  = blockIdx.x * (blockDim.x >> 5) + (threadIdx.x >> 5);
    int lane = threadIdx.x & 31;
    if (wid >= nrows) return;
    long r = idx[(long)wid * idx_stride];
    float a0 = src[r * D + lane], a1 = src[r * D + lane + 32];
    float ss = a0 * a0 + a1 * a1;
    #pragma unroll
    for (int off = 16; off; off >>= 1) ss += __shfl_xor(ss, off, 32);
    float inv = 1.0f / fmaxf(sqrtf(ss), 1e-12f);
    dst[(long)wid * D + lane]      = (__bf16)(a0 * inv);
    dst[(long)wid * D + lane + 32] = (__bf16)(a1 * inv);
}

// ---------------- S = X @ Y^T via v_wmma_f32_16x16x32_bf16, 1 wave per 16x16 tile ----
__global__ __launch_bounds__(128) void wmma_sim_kernel(const __bf16* __restrict__ X,
                                                       const __bf16* __restrict__ Y,
                                                       float* __restrict__ S, int n) {
    int wid    = blockIdx.x * (blockDim.x >> 5) + (threadIdx.x >> 5);
    int ntiles = n >> 4;
    int tm = wid / ntiles, tn = wid % ntiles;
    int lane = threadIdx.x & 31;
    int mr   = lane & 15;   // row (A) / col (B) within tile
    int half = lane >> 4;   // K half selector per ISA 16-bit fragment layout
    const __bf16* xrow = X + (long)(tm * 16 + mr) * D;
    const __bf16* yrow = Y + (long)(tn * 16 + mr) * D;
    v8f c = {};
    #pragma unroll
    for (int ko = 0; ko < D; ko += 32) {
        v16bf a, b;
        #pragma unroll
        for (int g = 0; g < 8; ++g) {
            // ISA 7.12.2: VGPR g holds K pair at ((g>=4)?16:0) + (g%4)*2 + half*8
            int kb = ((g & 4) << 2) + ((g & 3) << 1) + (half << 3) + ko;
            a[2 * g]     = xrow[kb];
            a[2 * g + 1] = xrow[kb + 1];
            b[2 * g]     = yrow[kb];
            b[2 * g + 1] = yrow[kb + 1];
        }
        c = __builtin_amdgcn_wmma_f32_16x16x32_bf16(
                /*neg_a=*/false, a, /*neg_b=*/false, b,
                /*c_mod=*/(short)0, c, /*reuse_a=*/false, /*reuse_b=*/false);
    }
    #pragma unroll
    for (int g = 0; g < 8; ++g) {
        int row = tm * 16 + g + 8 * half;   // C/D layout: VGPR g -> M = g (+8 for lanes 16-31)
        int col = tn * 16 + mr;
        S[(long)row * n + col] = c[g];
    }
}

// ---------------- row logsumexp(S/temp) minus diag/temp, accumulate weighted -> out ----
__global__ __launch_bounds__(128) void lse_kernel(const float* __restrict__ S,
                                                  float* __restrict__ out, int n, float weight) {
    int wid  = blockIdx.x * (blockDim.x >> 5) + (threadIdx.x >> 5);
    int lane = threadIdx.x & 31;
    if (wid >= n) return;
    const float* rowp = S + (long)wid * n;
    float m = -1e30f, s = 0.0f;
    for (int j = lane; j < n; j += 32) {
        float x = rowp[j] * 4.0f;          // 1 / C_TEMP = 4
        if (x > m) { s = s * expf(m - x) + 1.0f; m = x; }
        else       { s += expf(x - m); }
    }
    #pragma unroll
    for (int off = 16; off; off >>= 1) {
        float m2 = __shfl_xor(m, off, 32);
        float s2 = __shfl_xor(s, off, 32);
        float M  = fmaxf(m, m2);
        s = s * expf(m - M) + s2 * expf(m2 - M);
        m = M;
    }
    if (lane == 0) {
        float lse = m + logf(s);
        float pos = rowp[wid] * 4.0f;      // diagonal = <X_i, Y_i>
        atomicAdd(out, weight * (lse - pos));
    }
}

extern "C" void kernel_launch(void* const* d_in, const int* in_sizes, int n_in,
                              void* d_out, int out_size, void* d_ws, size_t ws_size,
                              hipStream_t stream) {
    const float* users_f   = (const float*)d_in[0];
    const float* items_f   = (const float*)d_in[1];
    const float* bundles_f = (const float*)d_in[2];
    const int*   il_row = (const int*)d_in[3];
    const int*   il_col = (const int*)d_in[4];
    const float* il_val = (const float*)d_in[5];
    const int*   bl_row = (const int*)d_in[6];
    const int*   bl_col = (const int*)d_in[7];
    const float* bl_val = (const float*)d_in[8];
    const int*   agg_row = (const int*)d_in[9];
    const int*   agg_col = (const int*)d_in[10];
    const float* agg_val = (const float*)d_in[11];
    const int*   users   = (const int*)d_in[12];
    const int*   bundles = (const int*)d_in[13];

    const long NU = in_sizes[0] / D, NI = in_sizes[1] / D, NB = in_sizes[2] / D;
    const long E_il = in_sizes[3], E_bl = in_sizes[6], E_agg = in_sizes[9];
    const int  batch = in_sizes[12];
    const long n_il = NU + NI, n_bl = NU + NB;

    // ---- workspace carve (256B aligned) ----
    char* ws = (char*)d_ws;
    size_t off = 0;
    auto carve = [&](size_t bytes) -> char* {
        char* p = ws + off;
        off += (bytes + 255) & ~(size_t)255;
        return p;
    };
    float*  F0     = (float*)carve(n_il * D * sizeof(float));
    float*  F1     = (float*)carve(n_il * D * sizeof(float));
    float*  IL_ACC = (float*)carve(n_il * D * sizeof(float));
    float*  BL_ACC = (float*)carve(n_bl * D * sizeof(float));
    float*  IL_B   = (float*)carve(NB * D * sizeof(float));
    __bf16* XB     = (__bf16*)carve((size_t)batch * D * sizeof(__bf16));
    __bf16* YB     = (__bf16*)carve((size_t)batch * D * sizeof(__bf16));
    float*  S      = (float*)carve((size_t)batch * batch * sizeof(float));
    float*  PRED   = (float*)carve((size_t)batch * 2 * sizeof(float));
    float*  out    = (float*)d_out;

    auto fillz = [&](float* p, long n) {
        long n4 = n >> 2;   // all our fills are multiples of 4 elements
        fillz_kernel<<<(int)((n4 + 255) / 256), 256, 0, stream>>>((float4*)p, n4);
    };
    auto spmm = [&](const int* r, const int* c, const float* v, const float* feat,
                    float* o, long nE, float scale) {
        long total = nE << 4;   // 16 threads per edge (float4 each)
        spmm_kernel<<<(int)((total + 255) / 256), 256, 0, stream>>>(r, c, v, feat, o, nE, scale);
    };
    auto acc_norm = [&](float* acc, const float* f, long nrows) {
        acc_norm_kernel<<<(int)((nrows + 7) / 8), 256, 0, stream>>>(acc, f, nrows);
    };

    // ===== item-level propagation over (U+I) graph =====
    hipMemcpyAsync(F0,          users_f, NU * D * sizeof(float), hipMemcpyDeviceToDevice, stream);
    hipMemcpyAsync(F0 + NU * D, items_f, NI * D * sizeof(float), hipMemcpyDeviceToDevice, stream);
    hipMemcpyAsync(IL_ACC, F0, n_il * D * sizeof(float), hipMemcpyDeviceToDevice, stream);
    fillz(F1, n_il * D);
    spmm(il_row, il_col, il_val, F0, F1, E_il, 0.5f);          // layer 0: /2
    acc_norm(IL_ACC, F1, n_il);
    fillz(F0, n_il * D);
    spmm(il_row, il_col, il_val, F1, F0, E_il, 1.0f / 3.0f);   // layer 1: /3
    acc_norm(IL_ACC, F0, n_il);

    // ===== bundle rep from item-level =====
    fillz(IL_B, NB * D);
    spmm(agg_row, agg_col, agg_val, IL_ACC + NU * D, IL_B, E_agg, 1.0f);

    // ===== bundle-level propagation over (U+B) graph =====
    hipMemcpyAsync(F0,          users_f,   NU * D * sizeof(float), hipMemcpyDeviceToDevice, stream);
    hipMemcpyAsync(F0 + NU * D, bundles_f, NB * D * sizeof(float), hipMemcpyDeviceToDevice, stream);
    hipMemcpyAsync(BL_ACC, F0, n_bl * D * sizeof(float), hipMemcpyDeviceToDevice, stream);
    fillz(F1, n_bl * D);
    spmm(bl_row, bl_col, bl_val, F0, F1, E_bl, 0.5f);
    acc_norm(BL_ACC, F1, n_bl);
    fillz(F0, n_bl * D);
    spmm(bl_row, bl_col, bl_val, F1, F0, E_bl, 1.0f / 3.0f);
    acc_norm(BL_ACC, F0, n_bl);

    // ===== BPR loss -> out[0], contrastive -> out[1] =====
    zero2_kernel<<<1, 32, 0, stream>>>(out);
    pred_kernel<<<(batch * 2 + 3) / 4, 128, 0, stream>>>(
        IL_ACC, IL_B, BL_ACC, BL_ACC + NU * D, users, bundles, PRED, batch);
    bpr_kernel<<<(batch + 255) / 256, 256, 0, stream>>>(PRED, out, batch);

    const int ntiles = batch / 16;                 // 128
    const int simBlocks = (ntiles * ntiles) / 4;   // 4 waves/block
    const int gnBlocks  = (batch + 7) / 8;
    const int lseBlocks = (batch + 3) / 4;
    const float w = 0.5f / (float)batch;

    // c_loss 1: X = norm(IL_u[users]), Y = norm(BL_u[users])
    gather_norm_bf16_kernel<<<gnBlocks, 256, 0, stream>>>(IL_ACC, users, 1, XB, batch);
    gather_norm_bf16_kernel<<<gnBlocks, 256, 0, stream>>>(BL_ACC, users, 1, YB, batch);
    wmma_sim_kernel<<<simBlocks, 128, 0, stream>>>(XB, YB, S, batch);
    lse_kernel<<<lseBlocks, 128, 0, stream>>>(S, out + 1, batch, w);

    // c_loss 2: X = norm(IL_b[bundles[:,0]]), Y = norm(BL_b[bundles[:,0]])
    gather_norm_bf16_kernel<<<gnBlocks, 256, 0, stream>>>(IL_B, bundles, 2, XB, batch);
    gather_norm_bf16_kernel<<<gnBlocks, 256, 0, stream>>>(BL_ACC + NU * D, bundles, 2, YB, batch);
    wmma_sim_kernel<<<simBlocks, 128, 0, stream>>>(XB, YB, S, batch);
    lse_kernel<<<lseBlocks, 128, 0, stream>>>(S, out + 1, batch, w);

    (void)n_in; (void)out_size; (void)ws_size;
}
